// RCNN_54726473286045
// MI455X (gfx1250) — matrix operand order repaired and checked by reading, not compiled, not run on hardware
//
#include <hip/hip_runtime.h>
#include <hip/hip_bf16.h>
#include <math.h>

typedef __attribute__((ext_vector_type(16))) __bf16 v16bf;
typedef __attribute__((ext_vector_type(8)))  float  v8f;
typedef __attribute__((ext_vector_type(4)))  unsigned int u32x4;

#define B_   256
#define T_   512
#define V_   50000
#define E_   128
#define H_   256
#define C_   128
#define D_   640   // 2H + E

__device__ __forceinline__ v8f wmma_bf16(v16bf a, v16bf b, v8f c) {
  return __builtin_amdgcn_wmma_f32_16x16x32_bf16(false, a, false, b, (short)0, c,
                                                 false, false);
}

// A-fragment (16x32 bf16, this lane's 16 elements): two 16B chunks 32B apart.
// lanes 0-15: K = kt*32 + {0..7, 16..23}; lanes 16-31: +8  (ISA 7.12.2 layout)
__device__ __forceinline__ v16bf load_a_frag(const __bf16* p) {
  union { v16bf v; u32x4 u[2]; } t;
  t.u[0] = *(const u32x4*)(p);
  t.u[1] = *(const u32x4*)(p + 16);
  return t.v;
}

__device__ __forceinline__ void atomic_max_f32(float* addr, float v) {
  if (v >= 0.0f) atomicMax((int*)addr, __float_as_int(v));
  else           atomicMin((unsigned int*)addr, __float_as_uint(v));
}

// CDNA5 async memory->LDS copy (ISA 15.18.3 op 98, tracked by ASYNCcnt).
// ldsOff = wave-relative LDS byte address (low 32 bits of generic pointer).
__device__ __forceinline__ void async_copy_b128(unsigned ldsOff,
                                                unsigned long long gaddr) {
  asm volatile("global_load_async_to_lds_b128 %0, %1, off"
               :: "v"(ldsOff), "v"(gaddr) : "memory");
}
__device__ __forceinline__ void wait_asynccnt0() {
  asm volatile("s_wait_asynccnt 0x0" ::: "memory");
}

// ---------------- prep kernels ----------------

__global__ void cvt_bf16_kernel(const float* __restrict__ src,
                                __bf16* __restrict__ dst, int n) {
  int i = blockIdx.x * blockDim.x + threadIdx.x;
  if (i < n) dst[i] = (__bf16)src[i];
}

// Pack a [K,N] f32 row-major matrix into the WMMA B-operand layout:
// dst[((kt*(N/16)+nt)*32 + lane)*16 + e] = src[(kt*32 + (e&7) + (lane>=16)*8 + (e>>3)*16)*N + nt*16 + lane%16]
__global__ void pack_b_kernel(const float* __restrict__ src,
                              __bf16* __restrict__ dst, int K, int N) {
  int gid = blockIdx.x * blockDim.x + threadIdx.x;
  if (gid >= K * N) return;
  int e    =  gid       & 15;
  int lane = (gid >> 4) & 31;
  int tile =  gid >> 9;
  int ntiles = N >> 4;
  int kt = tile / ntiles, nt = tile - kt * ntiles;
  int k = kt * 32 + (e & 7) + ((lane >> 4) << 3) + ((e >> 3) << 4);
  int n = nt * 16 + (lane & 15);
  dst[gid] = (__bf16)src[k * N + n];
}

// Gather e_cur straight into the middle columns [H, H+E) of X[B*T, 640]
__global__ void gather_ecur_kernel(const __bf16* __restrict__ embq,
                                   const int* __restrict__ idx,
                                   __bf16* __restrict__ X) {
  int gid = blockIdx.x * blockDim.x + threadIdx.x;   // B*T*16 threads
  int m = gid >> 4, c = gid & 15;
  const u32x4* src = (const u32x4*)(embq + (size_t)idx[m] * E_);
  u32x4* dst = (u32x4*)(X + (size_t)m * D_ + H_);
  dst[c] = src[c];
}

__global__ void init_p_kernel(float* __restrict__ P) {
  int i = blockIdx.x * blockDim.x + threadIdx.x;
  if (i < B_ * C_) P[i] = -2.0f;                     // < min(tanh)
}

// ---------------- recurrence ----------------
// 32 workgroups: blocks 0..15 = forward RNN (rows b0..b0+15), 16..31 = backward.
// Per step: acc = bias; acc += E_tile@W (4 WMMA); acc += h@U (8 WMMA); tanh;
// double-buffered h in LDS, one barrier per step, coalesced bf16 store into X.
__global__ __launch_bounds__(256) void rnn_kernel(
    const __bf16* __restrict__ embq,
    const int* __restrict__ idxL, const int* __restrict__ idxR,
    const __bf16* __restrict__ Wf_pk, const __bf16* __restrict__ Uf_pk,
    const float* __restrict__ bf_,
    const __bf16* __restrict__ Wb_pk, const __bf16* __restrict__ Ub_pk,
    const float* __restrict__ bb_,
    __bf16* __restrict__ X)
{
  extern __shared__ char smem[];
  __bf16* sU = (__bf16*)smem;                       // 131072 B packed U
  __bf16* sW = (__bf16*)(smem + 131072);            //  65536 B packed W
  __bf16* sH = (__bf16*)(smem + 131072 + 65536);    //  2 x 8192 B h double-buffer

  const int dir  = blockIdx.x >> 4;
  const int b0   = (blockIdx.x & 15) << 4;
  const int tid  = threadIdx.x;
  const int lane = tid & 31;
  const int wave = tid >> 5;
  const int mrow = lane & 15;
  const int hi   = lane >> 4;

  const int*    idx  = dir ? idxR  : idxL;
  const __bf16* Wp   = dir ? Wb_pk : Wf_pk;
  const __bf16* Up   = dir ? Ub_pk : Uf_pk;
  const float*  bias = dir ? bb_   : bf_;
  const int     xcol0 = dir ? (H_ + E_) : 0;

  { // async DMA of packed weights into LDS (ASYNCcnt path), clear h0
    unsigned ldsU = (unsigned)(uintptr_t)(void*)sU + (unsigned)tid * 16u;
    unsigned long long gaU = (unsigned long long)(uintptr_t)Up +
                             (unsigned long long)tid * 16ull;
    for (int i = tid; i < 131072 / 16; i += 256) {   // 32 async ops / lane
      async_copy_b128(ldsU, gaU);
      ldsU += 4096u; gaU += 4096ull;
    }
    unsigned ldsW = (unsigned)(uintptr_t)(void*)sW + (unsigned)tid * 16u;
    unsigned long long gaW = (unsigned long long)(uintptr_t)Wp +
                             (unsigned long long)tid * 16ull;
    for (int i = tid; i < 65536 / 16; i += 256) {    // 16 async ops / lane
      async_copy_b128(ldsW, gaW);
      ldsW += 4096u; gaW += 4096ull;
    }
    u32x4 z = {0u, 0u, 0u, 0u};
    u32x4* lH = (u32x4*)sH;
    for (int i = tid; i < 16384 / 16; i += 256) lH[i] = z;
    wait_asynccnt0();
  }
  __syncthreads();

  const int nt0 = wave, nt1 = wave + 8;              // 8 waves x 2 N-tiles = 16
  const float bias0 = bias[nt0 * 16 + mrow];
  const float bias1 = bias[nt1 * 16 + mrow];

  for (int step = 0; step < T_; ++step) {
    const int tt = dir ? (T_ - 1 - step) : step;
    __bf16* hCur = sH + (step & 1) * (16 * H_);
    __bf16* hNxt = sH + ((step + 1) & 1) * (16 * H_);

    const int token = idx[(b0 + mrow) * T_ + tt];
    const __bf16* erow = embq + (size_t)token * E_;
    if (step + 1 < T_) {
      const int tok2 = idx[(b0 + mrow) * T_ + (dir ? tt - 1 : tt + 1)];
      __builtin_prefetch(embq + (size_t)tok2 * E_, 0, 1);
    }

    v8f acc0, acc1;
#pragma unroll
    for (int r = 0; r < 8; ++r) { acc0[r] = bias0; acc1[r] = bias1; }

#pragma unroll
    for (int kt = 0; kt < E_ / 32; ++kt) {           // x_t @ W  (4 K-tiles)
      v16bf a  = load_a_frag(erow + kt * 32 + hi * 8);
      v16bf w0 = *(const v16bf*)(sW + ((kt * 16 + nt0) * 32 + lane) * 16);
      acc0 = wmma_bf16(a, w0, acc0);
      v16bf w1 = *(const v16bf*)(sW + ((kt * 16 + nt1) * 32 + lane) * 16);
      acc1 = wmma_bf16(a, w1, acc1);
    }
#pragma unroll
    for (int kt = 0; kt < H_ / 32; ++kt) {           // h_{t-1} @ U  (8 K-tiles)
      v16bf a  = load_a_frag(hCur + mrow * H_ + kt * 32 + hi * 8);
      v16bf u0 = *(const v16bf*)(sU + ((kt * 16 + nt0) * 32 + lane) * 16);
      acc0 = wmma_bf16(a, u0, acc0);
      v16bf u1 = *(const v16bf*)(sU + ((kt * 16 + nt1) * 32 + lane) * 16);
      acc1 = wmma_bf16(a, u1, acc1);
    }

#pragma unroll
    for (int r = 0; r < 8; ++r) {                    // tanh, scatter to h_next
      const int m = r + hi * 8;                      // C/D layout: lanes>=16 -> M+8
      hNxt[m * H_ + nt0 * 16 + mrow] = (__bf16)tanhf(acc0[r]);
      hNxt[m * H_ + nt1 * 16 + mrow] = (__bf16)tanhf(acc1[r]);
    }
    __syncthreads();

    { // coalesced store h(t) -> X[b, tt, xcol0 : xcol0+H]
      const int row = tid >> 4;
      const int ch  = tid & 15;
      const u32x4* src = (const u32x4*)(hNxt + row * H_);
      u32x4* dst = (u32x4*)(X + ((size_t)(b0 + row) * T_ + tt) * D_ + xcol0);
      dst[ch]      = src[ch];
      dst[ch + 16] = src[ch + 16];
    }
  }
}

// ---------------- Wc projection + tanh + max-over-T ----------------
// One block = one 16-row M-tile (16 consecutive t of one b); 8 waves = 8 N-tiles.
__global__ __launch_bounds__(256) void conv_max_kernel(
    const __bf16* __restrict__ X, const __bf16* __restrict__ Wc_pk,
    const float* __restrict__ bc, float* __restrict__ P)
{
  const int mt   = blockIdx.x;                       // 0..8191
  const int lane = threadIdx.x & 31;
  const int nt   = threadIdx.x >> 5;                 // 0..7
  const int mrow = lane & 15;
  const int hi   = lane >> 4;

  const __bf16* arow = X + ((size_t)mt * 16 + mrow) * D_;
  const float bias = bc[nt * 16 + mrow];
  v8f acc;
#pragma unroll
  for (int r = 0; r < 8; ++r) acc[r] = bias;

#pragma unroll
  for (int kt = 0; kt < D_ / 32; ++kt) {             // K = 640 -> 20 WMMAs
    v16bf a = load_a_frag(arow + kt * 32 + hi * 8);
    v16bf b = *(const v16bf*)(Wc_pk + ((kt * 8 + nt) * 32 + lane) * 16);
    acc = wmma_bf16(a, b, acc);
  }

  float m = -2.0f;
#pragma unroll
  for (int r = 0; r < 8; ++r) m = fmaxf(m, tanhf(acc[r]));
  m = fmaxf(m, __shfl_xor(m, 16, 32));               // combine M halves (same N)
  if (lane < 16)
    atomic_max_f32(&P[(size_t)(mt >> 5) * C_ + nt * 16 + lane], m);
}

// ---------------- head: sigmoid(P @ Wd + bd) ----------------
__global__ void head_kernel(const float* __restrict__ P, const float* __restrict__ Wd,
                            const float* __restrict__ bd, float* __restrict__ out)
{
  int b = blockIdx.x * blockDim.x + threadIdx.x;
  if (b >= B_) return;
  float a0 = bd[0], a1 = bd[1];
  for (int c = 0; c < C_; ++c) {
    float p = P[b * C_ + c];
    a0 += p * Wd[c * 2 + 0];
    a1 += p * Wd[c * 2 + 1];
  }
  out[b * 2 + 0] = 1.0f / (1.0f + expf(-a0));
  out[b * 2 + 1] = 1.0f / (1.0f + expf(-a1));
}

extern "C" void kernel_launch(void* const* d_in, const int* in_sizes, int n_in,
                              void* d_out, int out_size, void* d_ws, size_t ws_size,
                              hipStream_t stream)
{
  const int*   idxC = (const int*)  d_in[0];
  const int*   idxL = (const int*)  d_in[1];
  const int*   idxR = (const int*)  d_in[2];
  const float* emb  = (const float*)d_in[3];
  const float* Wf   = (const float*)d_in[4];
  const float* Uf   = (const float*)d_in[5];
  const float* bf_  = (const float*)d_in[6];
  const float* Wb   = (const float*)d_in[7];
  const float* Ub   = (const float*)d_in[8];
  const float* bb_  = (const float*)d_in[9];
  const float* Wc   = (const float*)d_in[10];
  const float* bc   = (const float*)d_in[11];
  const float* Wd   = (const float*)d_in[12];
  const float* bd   = (const float*)d_in[13];
  float* out = (float*)d_out;
  (void)in_sizes; (void)n_in; (void)out_size; (void)ws_size;

  char* ws = (char*)d_ws;
  size_t off = 0;
  auto carve = [&](size_t bytes) {
    char* p = ws + off;
    off = (off + bytes + 255) & ~(size_t)255;
    return p;
  };

  __bf16* embq  = (__bf16*)carve((size_t)V_ * E_ * 2);   // 12.8 MB
  __bf16* WfP   = (__bf16*)carve((size_t)E_ * H_ * 2);
  __bf16* WbP   = (__bf16*)carve((size_t)E_ * H_ * 2);
  __bf16* UfP   = (__bf16*)carve((size_t)H_ * H_ * 2);
  __bf16* UbP   = (__bf16*)carve((size_t)H_ * H_ * 2);
  __bf16* WcP   = (__bf16*)carve((size_t)D_ * C_ * 2);
  __bf16* X     = (__bf16*)carve((size_t)B_ * T_ * D_ * 2);  // 160 MB
  float*  Ppool = (float*) carve((size_t)B_ * C_ * 4);

  cvt_bf16_kernel<<<(V_ * E_ + 255) / 256, 256, 0, stream>>>(emb, embq, V_ * E_);
  pack_b_kernel<<<(E_ * H_ + 255) / 256, 256, 0, stream>>>(Wf, WfP, E_, H_);
  pack_b_kernel<<<(E_ * H_ + 255) / 256, 256, 0, stream>>>(Wb, WbP, E_, H_);
  pack_b_kernel<<<(H_ * H_ + 255) / 256, 256, 0, stream>>>(Uf, UfP, H_, H_);
  pack_b_kernel<<<(H_ * H_ + 255) / 256, 256, 0, stream>>>(Ub, UbP, H_, H_);
  pack_b_kernel<<<(D_ * C_ + 255) / 256, 256, 0, stream>>>(Wc, WcP, D_, C_);
  gather_ecur_kernel<<<(B_ * T_ * 16) / 256, 256, 0, stream>>>(embq, idxC, X);
  init_p_kernel<<<(B_ * C_ + 255) / 256, 256, 0, stream>>>(Ppool);

  rnn_kernel<<<32, 256, 131072 + 65536 + 16384, stream>>>(
      embq, idxL, idxR, WfP, UfP, bf_, WbP, UbP, bb_, X);

  conv_max_kernel<<<(B_ * T_) / 16, 256, 0, stream>>>(X, WcP, bc, Ppool);
  head_kernel<<<1, 256, 0, stream>>>(Ppool, Wd, bd, out);
}
